// EvidencePooling_82248623718961
// MI455X (gfx1250) — matrix operand order repaired and checked by reading, not compiled, not run on hardware
//
#include <hip/hip_runtime.h>
#include <hip/hip_bf16.h>
#include <math.h>

typedef __attribute__((ext_vector_type(16))) _Float16 v16h;
typedef __attribute__((ext_vector_type(8)))  float    v8f;

#define BATCH 16
#define NPIX (512*512)
#define NBINS 256
#define BPB 64                       // blocks per batch, pass1
#define TPB 256
#define PIX_PER_BLOCK (NPIX / BPB)   // 4096
#define THRESH 0.25f

// d_ws layout (bytes)
#define WS_SUMS 0                                   // float [B][8]: cnt,sumP0..3,sumSev,dmg,high
#define WS_MAXB (WS_SUMS + BATCH*8*4)               // uint  [B][8]: maxP0..3,maxSev (f32 bits, >=0)
#define WS_HCNT (WS_MAXB + BATCH*8*4)               // float [B][5][NBINS]
#define WS_HSUM (WS_HCNT + BATCH*5*NBINS*4)         // float [B][5][NBINS]
#define WS_A1   (WS_HSUM + BATCH*5*NBINS*4)         // _Float16 [16][32] normalized stats, K-padded
#define WS_TOTAL (WS_A1 + 16*32*2)

// out layout (floats): stats[16*18] | projected[16*256] | damaged[16] | high[16] | tar[16]
#define OUT_PROJ 288
#define OUT_DMG  4384
#define OUT_HIGH 4400
#define OUT_TAR  4416

__global__ void zero_kernel(unsigned* __restrict__ p, int n) {
    int i = blockIdx.x * blockDim.x + threadIdx.x;
    if (i < n) p[i] = 0u;
}

__device__ __forceinline__ float waveSum(float v) {
    v += __shfl_down(v, 16, 32); v += __shfl_down(v, 8, 32);
    v += __shfl_down(v, 4, 32);  v += __shfl_down(v, 2, 32);
    v += __shfl_down(v, 1, 32);  return v;
}
__device__ __forceinline__ float waveMax(float v) {
    v = fmaxf(v, __shfl_down(v, 16, 32)); v = fmaxf(v, __shfl_down(v, 8, 32));
    v = fmaxf(v, __shfl_down(v, 4, 32));  v = fmaxf(v, __shfl_down(v, 2, 32));
    v = fmaxf(v, __shfl_down(v, 1, 32));  return v;
}

// ---------------- Pass 1: stream 100 MB, softmax/sigmoid, masked reductions, LDS histograms ----
__global__ __launch_bounds__(TPB)
void pass1_kernel(const float* __restrict__ logits, const float* __restrict__ sevm,
                  const float* __restrict__ mask, float* __restrict__ ws)
{
    __shared__ float hC[5 * NBINS];
    __shared__ float hS[5 * NBINS];
    for (int i = threadIdx.x; i < 5 * NBINS; i += TPB) { hC[i] = 0.f; hS[i] = 0.f; }
    __syncthreads();

    const int b    = blockIdx.x / BPB;
    const int blk  = blockIdx.x % BPB;
    const int base = blk * PIX_PER_BLOCK;
    const float* lg = logits + (size_t)b * 4 * NPIX;
    const float* sv = sevm   + (size_t)b * NPIX;
    const float* mk = mask   + (size_t)b * NPIX;

    float cnt=0.f, sp0=0.f, sp1=0.f, sp2=0.f, sp3=0.f, ssev=0.f, dmg=0.f, hig=0.f;
    float mp0=0.f, mp1=0.f, mp2=0.f, mp3=0.f, msev=0.f;

    for (int it = 0; it < PIX_PER_BLOCK / (TPB * 4); ++it) {
        const int n = base + (it * TPB + threadIdx.x) * 4;
        float4 e0 = *(const float4*)(lg + n);
        float4 e1 = *(const float4*)(lg + NPIX + n);
        float4 e2 = *(const float4*)(lg + 2 * NPIX + n);
        float4 e3 = *(const float4*)(lg + 3 * NPIX + n);
        float4 s4 = *(const float4*)(sv + n);
        float4 m4 = *(const float4*)(mk + n);
        float A0[4] = {e0.x, e0.y, e0.z, e0.w};
        float A1[4] = {e1.x, e1.y, e1.z, e1.w};
        float A2[4] = {e2.x, e2.y, e2.z, e2.w};
        float A3[4] = {e3.x, e3.y, e3.z, e3.w};
        float S4[4] = {s4.x, s4.y, s4.z, s4.w};
        float M4[4] = {m4.x, m4.y, m4.z, m4.w};
        #pragma unroll
        for (int j = 0; j < 4; ++j) {
            float v0 = A0[j], v1 = A1[j], v2 = A2[j], v3 = A3[j];
            float mx = fmaxf(fmaxf(v0, v1), fmaxf(v2, v3));
            float x0 = __expf(v0 - mx), x1 = __expf(v1 - mx);
            float x2 = __expf(v2 - mx), x3 = __expf(v3 - mx);
            float inv = 1.f / (x0 + x1 + x2 + x3);
            float p0 = x0 * inv, p1 = x1 * inv, p2 = x2 * inv, p3 = x3 * inv;
            float se = 1.f / (1.f + __expf(-S4[j]));
            if (M4[j] > 0.5f) {
                cnt += 1.f;
                sp0 += p0; sp1 += p1; sp2 += p2; sp3 += p3; ssev += se;
                mp0 = fmaxf(mp0, p0); mp1 = fmaxf(mp1, p1);
                mp2 = fmaxf(mp2, p2); mp3 = fmaxf(mp3, p3);
                msev = fmaxf(msev, se);
                dmg += (p1 + p2 + p3 > THRESH) ? 1.f : 0.f;
                hig += (p2 + p3 > THRESH) ? 1.f : 0.f;
                float vals[5] = {p0, p1, p2, p3, se};
                #pragma unroll
                for (int ch = 0; ch < 5; ++ch) {
                    float v = vals[ch];
                    int bin = (int)(v * (float)NBINS);
                    bin = bin > NBINS - 1 ? NBINS - 1 : (bin < 0 ? 0 : bin);
                    atomicAdd(&hC[ch * NBINS + bin], 1.f);
                    atomicAdd(&hS[ch * NBINS + bin], v);
                }
            }
        }
    }

    // wave reduce scalar accumulators -> global atomics
    float* SUMS = ws + WS_SUMS / 4 + b * 8;
    unsigned* MAXB = (unsigned*)ws + WS_MAXB / 4 + b * 8;
    float r[8] = {cnt, sp0, sp1, sp2, sp3, ssev, dmg, hig};
    float m[5] = {mp0, mp1, mp2, mp3, msev};
    const int lane = threadIdx.x & 31;
    #pragma unroll
    for (int i = 0; i < 8; ++i) {
        float v = waveSum(r[i]);
        if (lane == 0 && v != 0.f) atomicAdd(&SUMS[i], v);
    }
    #pragma unroll
    for (int i = 0; i < 5; ++i) {
        float v = waveMax(m[i]);
        if (lane == 0) atomicMax(&MAXB[i], __float_as_uint(v));
    }

    __syncthreads();
    float* gC = ws + WS_HCNT / 4 + (size_t)b * 5 * NBINS;
    float* gS = ws + WS_HSUM / 4 + (size_t)b * 5 * NBINS;
    for (int i = threadIdx.x; i < 5 * NBINS; i += TPB) {
        if (hC[i] != 0.f) { atomicAdd(&gC[i], hC[i]); atomicAdd(&gS[i], hS[i]); }
    }
}

// ---------------- Finalize: top-k from histograms, 18 stats, layernorm -> f16 A-matrix ---------
__global__ __launch_bounds__(64)
void finalize_kernel(float* __restrict__ ws, const float* __restrict__ ln_w,
                     const float* __restrict__ ln_b, float* __restrict__ out)
{
    const int b = blockIdx.x;
    float* SUMS = ws + WS_SUMS / 4 + b * 8;
    unsigned* MAXB = (unsigned*)ws + WS_MAXB / 4 + b * 8;
    const float* HC = ws + WS_HCNT / 4 + (size_t)b * 5 * NBINS;
    const float* HS = ws + WS_HSUM / 4 + (size_t)b * 5 * NBINS;
    _Float16* A1 = (_Float16*)((char*)ws + WS_A1) + b * 32;

    __shared__ float tkm[5];
    const float cnt = SUMS[0];
    const bool  has = cnt > 0.f;
    const float safe = fmaxf(cnt, 1.f);
    const float kk = fmaxf(1.f, roundf(cnt * 0.1f));

    const int t = threadIdx.x;
    if (t < 5) {  // top-k mean per channel: scan bins high->low, interpolate boundary bin
        const float* hc = HC + t * NBINS;
        const float* hs = HS + t * NBINS;
        float cacc = 0.f, sacc = 0.f;
        for (int bin = NBINS - 1; bin >= 0; --bin) {
            float c = hc[bin];
            if (c > 0.f) {
                float s = hs[bin];
                if (cacc + c >= kk) { sacc += s * ((kk - cacc) / c); break; }
                cacc += c; sacc += s;
            }
        }
        tkm[t] = has ? (sacc / kk) : 0.f;
    }
    __syncthreads();

    if (t == 0) {
        float st[18];
        st[0] = SUMS[1] / safe; st[1] = SUMS[2] / safe;
        st[2] = SUMS[3] / safe; st[3] = SUMS[4] / safe;
        st[4] = __uint_as_float(MAXB[0]); st[5] = __uint_as_float(MAXB[1]);
        st[6] = __uint_as_float(MAXB[2]); st[7] = __uint_as_float(MAXB[3]);
        st[8] = tkm[0]; st[9] = tkm[1]; st[10] = tkm[2]; st[11] = tkm[3];
        st[12] = SUMS[5] / safe;
        st[13] = __uint_as_float(MAXB[4]);
        st[14] = tkm[4];
        st[15] = SUMS[6] / safe;
        st[16] = SUMS[7] / safe;
        st[17] = has ? (cnt / (float)NPIX) : 0.f;

        float mu = 0.f;
        for (int j = 0; j < 18; ++j) mu += st[j];
        mu *= (1.f / 18.f);
        float var = 0.f;
        for (int j = 0; j < 18; ++j) { float d = st[j] - mu; var += d * d; }
        var *= (1.f / 18.f);
        float inv = rsqrtf(var + 1e-5f);

        for (int j = 0; j < 18; ++j) out[b * 18 + j] = st[j];
        out[OUT_DMG + b]  = st[15];
        out[OUT_HIGH + b] = st[16];
        out[OUT_TAR + b]  = st[17];
        for (int j = 0; j < 32; ++j) {
            float nv = (j < 18) ? ((st[j] - mu) * inv * ln_w[j] + ln_b[j]) : 0.f;
            A1[j] = (_Float16)nv;
        }
    }
}

// ---------------- MLP: (16x32)@(32x256) -> GELU -> (16x256)@(256x256), all via WMMA f16 -------
// ISA fragment layouts (wave32): A 16x32 f16: row m=lane&15; halves i: pair p=i/2,
// K = (p<4 ? hi*8+2p : 16+hi*8+2(p-4)) + (i&1), hi=(lane>=16). B 32x16 f16: col n=lane&15,
// K as for A; this code loads both A and B with K = hi*16 + i. C/D f32: n=lane&15, VGPR v -> M = v + hi*8.
__device__ __forceinline__ int a_koff(int hi, int i) {
    int p = i >> 1, r = i & 1;
    return (p < 4) ? (hi * 8 + 2 * p + r) : (16 + hi * 8 + 2 * (p - 4) + r);
}

__global__ __launch_bounds__(512)
void mlp_kernel(const float* __restrict__ w1, const float* __restrict__ b1,
                const float* __restrict__ w2, const float* __restrict__ b2v,
                const _Float16* __restrict__ A1, float* __restrict__ out)
{
    __shared__ _Float16 h1s[16 * 256];
    const int lane = threadIdx.x & 31;
    const int wv   = threadIdx.x >> 5;        // 16 waves -> 16 N-tiles of width 16
    const int hi   = (lane < 16) ? 0 : 1;
    const int mrow = lane & 15;
    const int coln = wv * 16 + (lane & 15);

    // GEMM1: normed(16x32,padded) @ w1(32x256,padded K)
    v16h a;
    #pragma unroll
    for (int i = 0; i < 16; ++i) a[i] = A1[mrow * 32 + hi * 16 + i];
    v16h bf;
    #pragma unroll
    for (int i = 0; i < 16; ++i) {
        int k = hi * 16 + i;
        bf[i] = (k < 18) ? (_Float16)w1[k * 256 + coln] : (_Float16)0.f;
    }
    v8f acc = {};
    acc = __builtin_amdgcn_wmma_f32_16x16x32_f16(false, a, false, bf, (short)0, acc, false, false);

    const float bias1 = b1[coln];
    #pragma unroll
    for (int v = 0; v < 8; ++v) {
        int m = v + hi * 8;
        float x = acc[v] + bias1;
        float g = 0.5f * x * (1.0f + erff(x * 0.70710678118654752f));  // exact GELU
        h1s[m * 256 + coln] = (_Float16)g;
    }
    __syncthreads();

    // GEMM2: h1(16x256) @ w2(256x256), 8 K-steps of 32
    v8f acc2 = {};
    #pragma unroll
    for (int kt = 0; kt < 8; ++kt) {
        const int kb = kt * 32;
        v16h a2, b2f;
        #pragma unroll
        for (int i = 0; i < 16; ++i) a2[i] = h1s[mrow * 256 + kb + hi * 16 + i];
        #pragma unroll
        for (int i = 0; i < 16; ++i) b2f[i] = (_Float16)w2[(kb + hi * 16 + i) * 256 + coln];
        acc2 = __builtin_amdgcn_wmma_f32_16x16x32_f16(false, a2, false, b2f, (short)0, acc2, false, false);
    }
    const float bias2 = b2v[coln];
    #pragma unroll
    for (int v = 0; v < 8; ++v) {
        int m = v + hi * 8;
        out[OUT_PROJ + m * 256 + coln] = acc2[v] + bias2;
    }
}

extern "C" void kernel_launch(void* const* d_in, const int* in_sizes, int n_in,
                              void* d_out, int out_size, void* d_ws, size_t ws_size,
                              hipStream_t stream) {
    const float* logits = (const float*)d_in[0];
    const float* sevm   = (const float*)d_in[1];
    const float* mask   = (const float*)d_in[2];
    const float* ln_w   = (const float*)d_in[3];
    const float* ln_b   = (const float*)d_in[4];
    const float* w1     = (const float*)d_in[5];
    const float* b1     = (const float*)d_in[6];
    const float* w2     = (const float*)d_in[7];
    const float* b2     = (const float*)d_in[8];
    float* out = (float*)d_out;
    float* ws  = (float*)d_ws;

    const int zwords = WS_TOTAL / 4;
    zero_kernel<<<(zwords + 255) / 256, 256, 0, stream>>>((unsigned*)d_ws, zwords);
    pass1_kernel<<<BATCH * BPB, TPB, 0, stream>>>(logits, sevm, mask, ws);
    finalize_kernel<<<BATCH, 64, 0, stream>>>(ws, ln_w, ln_b, out);
    mlp_kernel<<<1, 512, 0, stream>>>(w1, b1, w2, b2,
        (const _Float16*)((const char*)d_ws + WS_A1), out);
}
